// EqStftPBC_51436528337748
// MI455X (gfx1250) — matrix operand order repaired and checked by reading, not compiled, not run on hardware
//
#include <hip/hip_runtime.h>
#include <hip/hip_bf16.h>

namespace {

constexpr int   Bn   = 2;
constexpr int   Lx   = 2080;
constexpr int   Mm   = 2;
constexpr int   T    = 51;
constexpr int   F    = 80;
constexpr int   HOP  = 40;
constexpr int   R    = Bn * T * Mm;   // 204 spectra
constexpr int   Rp   = 208;           // padded rows: 13 * 16
constexpr int   Kp   = 96;            // padded K:    3 * 32
constexpr int   N2   = 40;            // n2 in [-20, 19]
constexpr int   CROP = 20;
constexpr int   Lout = Lx - 2 * CROP; // 2040
constexpr float PI_F = 3.14159265358979323846f;

typedef __attribute__((ext_vector_type(16))) _Float16 v16h;
typedef __attribute__((ext_vector_type(8)))  float    v8f;

// ---------------------------------------------------------------------------
// Stage 1: STFT + forward DFT (80-point, direct).  X[r][f], r = (b*T+t)*2+m
// ---------------------------------------------------------------------------
__global__ void k_dft(const float* __restrict__ xr, const float* __restrict__ xi,
                      float* __restrict__ Xre, float* __restrict__ Xim) {
  int idx = blockIdx.x * blockDim.x + threadIdx.x;
  if (idx >= R * F) return;
  int r = idx / F, f = idx % F;
  int b  = r / (Mm * T);
  int t  = (r / Mm) % T;
  int mi = r % Mm;
  const float w0 = -2.0f * PI_F * (float)f / (float)F;
  float sre = 0.0f, sim = 0.0f;
  for (int n = 0; n < F; ++n) {
    int l = t * HOP + n;
    float ar = xr[(b * Lx + l) * Mm + mi];
    float ai = xi[(b * Lx + l) * Mm + mi];
    float s, c;
    __sincosf(w0 * (float)n, &s, &c);     // e^{-2pi i f n / F} = c + i s
    sre += ar * c - ai * s;
    sim += ar * s + ai * c;
  }
  Xre[idx] = sre;
  Xim[idx] = sim;
}

// ---------------------------------------------------------------------------
// Stage 2: Z[n2][r][u] = X_t[u]*conj(X_t[(u-n2)%F]) + X_{t-1}[u]*conj(...)
// stored f16, zero-padded to Rp x Kp
// ---------------------------------------------------------------------------
__global__ void k_zbuild(const float* __restrict__ Xre, const float* __restrict__ Xim,
                         _Float16* __restrict__ Zre, _Float16* __restrict__ Zim) {
  int idx = blockIdx.x * blockDim.x + threadIdx.x;
  if (idx >= N2 * Rp * Kp) return;
  int n2i = idx / (Rp * Kp);
  int r   = (idx / Kp) % Rp;
  int u   = idx % Kp;
  if (r >= R || u >= F) { Zre[idx] = (_Float16)0.0f; Zim[idx] = (_Float16)0.0f; return; }
  int n2v = n2i - 20;
  int u2  = ((u - n2v) % F + F) % F;
  int b  = r / (Mm * T);
  int t  = (r / Mm) % T;
  int mi = r % Mm;
  int tp = (t + T - 1) % T;                 // roll(A, 1, axis=T)
  int rp = (b * T + tp) * Mm + mi;
  float ar = Xre[r * F + u],  ai = Xim[r * F + u];
  float br = Xre[r * F + u2], bi = Xim[r * F + u2];
  float zr = ar * br + ai * bi;             // X * conj(X2)
  float zi = ai * br - ar * bi;
  ar = Xre[rp * F + u];  ai = Xim[rp * F + u];
  br = Xre[rp * F + u2]; bi = Xim[rp * F + u2];
  zr += ar * br + ai * bi;
  zi += ai * br - ar * bi;
  Zre[idx] = (_Float16)zr;
  Zim[idx] = (_Float16)zi;
}

// ---------------------------------------------------------------------------
// Stage 3: banded weight matrices, stored TRANSPOSED for contiguous WMMA-B
// fragments: Wt[n2][f][u] = w[n1=(f-u) mod F, n2]  (f16, u padded to Kp)
// ---------------------------------------------------------------------------
__global__ void k_wbuild(const float* __restrict__ wre, const float* __restrict__ wim,
                         _Float16* __restrict__ Wt_re, _Float16* __restrict__ Wt_im) {
  int idx = blockIdx.x * blockDim.x + threadIdx.x;
  if (idx >= N2 * F * Kp) return;
  int n2i = idx / (F * Kp);
  int f   = (idx / Kp) % F;
  int u   = idx % Kp;
  float vr = 0.0f, vi = 0.0f;
  if (u < F) {
    int dd = ((f - u) % F + F) % F;
    int n1 = 100;                           // invalid
    if (dd <= 19)      n1 = dd;
    else if (dd >= 60) n1 = dd - F;         // [-20, -1]
    if (n1 >= -20 && n1 <= 19) {
      int k = (n1 + 20) * 40 + n2i;         // IDX enumeration order (all pairs kept)
      vr = wre[k];
      vi = wim[k];
    }
  }
  Wt_re[idx] = (_Float16)vr;
  Wt_im[idx] = (_Float16)vi;
}

// ---------------------------------------------------------------------------
// Stage 4: batched complex GEMM via WMMA: S[n2] = Z[n2] (Rp x Kp) @ Wm[n2] (Kp x F)
// One wave per (column-tile, n2); B fragments (full K=96, re+im) kept in
// registers and reused across all 13 row tiles. 4 real accumulators per tile
// (f16 WMMA forbids A/B NEG). S stored f-major -> each lane's 8 outputs are
// contiguous (32B-aligned) so stores merge into b128s.
// ---------------------------------------------------------------------------
__device__ __forceinline__ v16h load_a_tile(const _Float16* __restrict__ A, int lda,
                                            int rbase, int kbase, int lane) {
  // A fragment in this code's contiguous k order (matches B): lanes 0-15 -> K{0..15}; lanes 16-31 -> K{16..31}
  const int mrow  = lane & 15;
  const int khalf = (lane & 16) ? 16 : 0;
  const _Float16* p = A + (size_t)(rbase + mrow) * lda + kbase;
  v16h a;
#pragma unroll
  for (int j = 0; j < 16; ++j) a[j] = p[khalf + j];
  return a;
}

__global__ __launch_bounds__(32) void k_gemm(const _Float16* __restrict__ Zre,
                                             const _Float16* __restrict__ Zim,
                                             const _Float16* __restrict__ Wt_re,
                                             const _Float16* __restrict__ Wt_im,
                                             float* __restrict__ Sre,
                                             float* __restrict__ Sim) {
  const int lane  = threadIdx.x;
  const int cbase = blockIdx.x * 16;   // 5 tiles over F=80
  const int n2    = blockIdx.y;        // 40 batches

  const _Float16* Ar = Zre + (size_t)n2 * Rp * Kp;
  const _Float16* Ai = Zim + (size_t)n2 * Rp * Kp;

  // B fragment, same contiguous k order as A: element j of lane = K(kbase+koff+j), N = lane&15.
  // With transposed storage Wt[n2][f][u] each fragment is a contiguous 32B run.
  const int ncol = lane & 15;
  const int koff = (lane & 16) ? 16 : 0;
  const _Float16* Bp_re = Wt_re + ((size_t)n2 * F + (cbase + ncol)) * Kp;
  const _Float16* Bp_im = Wt_im + ((size_t)n2 * F + (cbase + ncol)) * Kp;

  v16h br[3], bi[3];
#pragma unroll
  for (int ks = 0; ks < 3; ++ks) {
#pragma unroll
    for (int j = 0; j < 16; ++j) {
      br[ks][j] = Bp_re[ks * 32 + koff + j];
      bi[ks][j] = Bp_im[ks * 32 + koff + j];
    }
  }

  const int mh = (lane & 16) ? 8 : 0;
  // f-major output row for this lane's column
  float* so_re = Sre + ((size_t)n2 * F + cbase + ncol) * Rp;
  float* so_im = Sim + ((size_t)n2 * F + cbase + ncol) * Rp;

  for (int rt = 0; rt < Rp / 16; ++rt) {
    const int rbase = rt * 16;
    if (rt + 1 < Rp / 16) {  // warm next row tile's A data into cache
      __builtin_prefetch(Ar + (size_t)(rbase + 16) * Kp, 0, 3);
      __builtin_prefetch(Ai + (size_t)(rbase + 16) * Kp, 0, 3);
    }
    v8f p1 = {}, p2 = {}, p3 = {}, p4 = {};
#pragma unroll
    for (int ks = 0; ks < 3; ++ks) {
      v16h ar = load_a_tile(Ar, Kp, rbase, ks * 32, lane);
      v16h ai = load_a_tile(Ai, Kp, rbase, ks * 32, lane);
      p1 = __builtin_amdgcn_wmma_f32_16x16x32_f16(false, ar, false, br[ks], (short)0, p1, false, false);
      p2 = __builtin_amdgcn_wmma_f32_16x16x32_f16(false, ai, false, bi[ks], (short)0, p2, false, false);
      p3 = __builtin_amdgcn_wmma_f32_16x16x32_f16(false, ar, false, bi[ks], (short)0, p3, false, false);
      p4 = __builtin_amdgcn_wmma_f32_16x16x32_f16(false, ai, false, br[ks], (short)0, p4, false, false);
    }
    // D layout: element i -> M = rbase+mh+i, N = ncol; f-major S makes the 8
    // values contiguous in memory (32B aligned: Rp and rbase+mh are /8).
    const size_t o = (size_t)rbase + mh;
#pragma unroll
    for (int i = 0; i < 8; ++i) so_re[o + i] = p1[i] - p2[i];  // re = Zre*Wre - Zim*Wim
#pragma unroll
    for (int i = 0; i < 8; ++i) so_im[o + i] = p3[i] + p4[i];  // im = Zre*Wim + Zim*Wre
  }
}

// ---------------------------------------------------------------------------
// Stage 5: delta[r][f] = bias + sum_{n2} X[r][(f-n2)%F] * S[n2][f][r]
// ---------------------------------------------------------------------------
__global__ void k_delta(const float* __restrict__ Xre, const float* __restrict__ Xim,
                        const float* __restrict__ Sre, const float* __restrict__ Sim,
                        const float* __restrict__ b_re, const float* __restrict__ b_im,
                        float* __restrict__ Dre, float* __restrict__ Dim) {
  int idx = blockIdx.x * blockDim.x + threadIdx.x;
  if (idx >= R * F) return;
  int r = idx / F, f = idx % F;
  float accr = b_re[0], acci = b_im[0];
  for (int n2i = 0; n2i < N2; ++n2i) {
    int n2v = n2i - 20;
    int f2  = ((f - n2v) % F + F) % F;
    float xr = Xre[r * F + f2], xi = Xim[r * F + f2];
    size_t so = ((size_t)n2i * F + f) * Rp + r;   // f-major S
    float sr = Sre[so], si = Sim[so];
    accr += xr * sr - xi * si;
    acci += xr * si + xi * sr;
  }
  Dre[idx] = accr;
  Dim[idx] = acci;
}

// ---------------------------------------------------------------------------
// Stage 6: inverse DFT of delta -> time-domain frames (includes 1/F)
// ---------------------------------------------------------------------------
__global__ void k_idft(const float* __restrict__ Dre, const float* __restrict__ Dim,
                       float* __restrict__ Yre, float* __restrict__ Yim) {
  int idx = blockIdx.x * blockDim.x + threadIdx.x;
  if (idx >= R * F) return;
  int r = idx / F, n = idx % F;
  const float w0 = 2.0f * PI_F * (float)n / (float)F;
  float sre = 0.0f, sim = 0.0f;
  for (int f = 0; f < F; ++f) {
    float s, c;
    __sincosf(w0 * (float)f, &s, &c);     // e^{+2pi i f n / F}
    float dr = Dre[r * F + f], di = Dim[r * F + f];
    sre += dr * c - di * s;
    sim += dr * s + di * c;
  }
  const float inv = 1.0f / (float)F;
  Yre[idx] = sre * inv;
  Yim[idx] = sim * inv;
}

// ---------------------------------------------------------------------------
// Stage 7: overlap-add (coverage-normalized), scale by P, add x, crop, emit
// complex64-interleaved output.
// ---------------------------------------------------------------------------
__global__ void k_out(const float* __restrict__ xr, const float* __restrict__ xi,
                      const float* __restrict__ task_info,
                      const float* __restrict__ Yre, const float* __restrict__ Yim,
                      float* __restrict__ out) {
  int idx = blockIdx.x * blockDim.x + threadIdx.x;
  if (idx >= Bn * Lout * Mm) return;
  int b  = idx / (Lout * Mm);
  int lp = (idx / Mm) % Lout;
  int mi = idx % Mm;
  int l  = lp + CROP;

  float dr = 0.0f, di = 0.0f;
  int cov = 0;
  int t0 = l / HOP;
  for (int t = t0 - 1; t <= t0; ++t) {
    if (t < 0 || t >= T) continue;
    int n = l - t * HOP;
    if (n < 0 || n >= F) continue;
    int r = (b * T + t) * Mm + mi;
    dr += Yre[r * F + n];
    di += Yim[r * F + n];
    ++cov;
  }
  float inv = 1.0f / (float)cov;
  dr *= inv; di *= inv;

  float P = __powf(10.0f, task_info[b * 4 + 0] * 0.1f) * 0.5f;  // /m, m=2
  float outr = xr[(b * Lx + l) * Mm + mi] + dr * P;
  float outi = xi[(b * Lx + l) * Mm + mi] + di * P;
  out[(size_t)idx * 2 + 0] = outr;
  out[(size_t)idx * 2 + 1] = outi;
}

} // anonymous namespace

extern "C" void kernel_launch(void* const* d_in, const int* in_sizes, int n_in,
                              void* d_out, int out_size, void* d_ws, size_t ws_size,
                              hipStream_t stream) {
  const float* xr  = (const float*)d_in[0];
  const float* xi  = (const float*)d_in[1];
  const float* ti  = (const float*)d_in[2];
  const float* wre = (const float*)d_in[3];
  const float* wim = (const float*)d_in[4];
  const float* bre = (const float*)d_in[5];
  const float* bim = (const float*)d_in[6];
  float* out = (float*)d_out;

  // Workspace carve-up (all offsets 256B aligned, ~10 MB total)
  char* ws = (char*)d_ws;
  size_t off = 0;
  auto carve = [&](size_t bytes) -> void* {
    void* p = ws + off;
    off = (off + bytes + 255) & ~(size_t)255;
    return p;
  };
  float*    Xre = (float*)   carve((size_t)R * F * sizeof(float));
  float*    Xim = (float*)   carve((size_t)R * F * sizeof(float));
  _Float16* Zre = (_Float16*)carve((size_t)N2 * Rp * Kp * sizeof(_Float16));
  _Float16* Zim = (_Float16*)carve((size_t)N2 * Rp * Kp * sizeof(_Float16));
  _Float16* Wtr = (_Float16*)carve((size_t)N2 * F * Kp * sizeof(_Float16));
  _Float16* Wti = (_Float16*)carve((size_t)N2 * F * Kp * sizeof(_Float16));
  float*    Sre = (float*)   carve((size_t)N2 * F * Rp * sizeof(float));
  float*    Sim = (float*)   carve((size_t)N2 * F * Rp * sizeof(float));
  float*    Dre = (float*)   carve((size_t)R * F * sizeof(float));
  float*    Dim = (float*)   carve((size_t)R * F * sizeof(float));
  float*    Yre = (float*)   carve((size_t)R * F * sizeof(float));
  float*    Yim = (float*)   carve((size_t)R * F * sizeof(float));

  const int BLK = 256;

  // 1) STFT + DFT
  k_dft<<<(R * F + BLK - 1) / BLK, BLK, 0, stream>>>(xr, xi, Xre, Xim);

  // 2) Z = Y_t + Y_{t-1} (f16, padded)
  k_zbuild<<<(N2 * Rp * Kp + BLK - 1) / BLK, BLK, 0, stream>>>(Xre, Xim, Zre, Zim);

  // 3) banded weight matrices, transposed layout (f16, padded)
  k_wbuild<<<(N2 * F * Kp + BLK - 1) / BLK, BLK, 0, stream>>>(wre, wim, Wtr, Wti);

  // 4) 40 batched complex GEMMs (208x80x96) on the WMMA path; B held in regs
  {
    dim3 grid(F / 16, N2);           // (5, 40) waves
    dim3 block(32, 1, 1);            // one wave per (col-tile, n2)
    k_gemm<<<grid, block, 0, stream>>>(Zre, Zim, Wtr, Wti, Sre, Sim);
  }

  // 5) contract over n2 with shifted X, add bias
  k_delta<<<(R * F + BLK - 1) / BLK, BLK, 0, stream>>>(Xre, Xim, Sre, Sim, bre, bim, Dre, Dim);

  // 6) inverse DFT
  k_idft<<<(R * F + BLK - 1) / BLK, BLK, 0, stream>>>(Dre, Dim, Yre, Yim);

  // 7) overlap-add + scale + crop + emit complex64
  k_out<<<(Bn * Lout * Mm + BLK - 1) / BLK, BLK, 0, stream>>>(xr, xi, ti, Yre, Yim, out);
}
